// StereoMatch_28733331210650
// MI455X (gfx1250) — compile-verified, matrix operand order checked by reading, not run on hardware
//
#include <hip/hip_runtime.h>
#include <hip/hip_bf16.h>
#include <math.h>

// Problem constants (match reference)
#define IW   960
#define IH   540
#define ICH  3
#define ND   60      // disparities 1..60
#define XT   16      // output cols per workgroup tile
#define YT   56      // output rows per workgroup tile
#define RT   64      // rows incl. 9-tap halo (YT + 8)
#define NWIN 243.0f  // 9*9*3
#define CEPS 1e-6f

typedef __attribute__((ext_vector_type(16))) _Float16 v16h;
typedef __attribute__((ext_vector_type(8)))  _Float16 v8h;
typedef __attribute__((ext_vector_type(8)))  float    v8f;
typedef __attribute__((ext_vector_type(4)))  unsigned int v4u;

// ---------------------------------------------------------------------------
// Stats pass A: horizontal 9-tap box sums of (sum_c v) and (sum_c v^2), L & R.
// Writes 4 planes [4][H][W] into rs.
// ---------------------------------------------------------------------------
__global__ __launch_bounds__(256) void stats_row_kernel(const float* __restrict__ L,
                                                        const float* __restrict__ R,
                                                        float* __restrict__ rs) {
  const int total = IH * IW;
  int idx = blockIdx.x * blockDim.x + threadIdx.x;
  if (idx >= total) return;
  int y = idx / IW, x = idx - y * IW;
  float sL = 0.f, sL2 = 0.f, sR = 0.f, sR2 = 0.f;
  #pragma unroll
  for (int dx = -4; dx <= 4; ++dx) {
    int xx = x + dx;
    if ((unsigned)xx < (unsigned)IW) {
      #pragma unroll
      for (int c = 0; c < ICH; ++c) {
        float vl = L[(c * IH + y) * IW + xx];
        float vr = R[(c * IH + y) * IW + xx];
        sL += vl; sL2 += vl * vl;
        sR += vr; sR2 += vr * vr;
      }
    }
  }
  rs[0 * total + idx] = sL;
  rs[1 * total + idx] = sL2;
  rs[2 * total + idx] = sR;
  rs[3 * total + idx] = sR2;
}

// ---------------------------------------------------------------------------
// Stats pass B: vertical 9-tap, then derive {LSum, sqrtL, RSum, sqrtR}.
// sqrtX = sqrt(X2Sum - XSum^2/n + 1e-5)  (algebraic simplification of ref)
// ---------------------------------------------------------------------------
__global__ __launch_bounds__(256) void stats_col_kernel(const float* __restrict__ rs,
                                                        float* __restrict__ st) {
  const int total = IH * IW;
  int idx = blockIdx.x * blockDim.x + threadIdx.x;
  if (idx >= total) return;
  int y = idx / IW, x = idx - y * IW;
  float sL = 0.f, sL2 = 0.f, sR = 0.f, sR2 = 0.f;
  #pragma unroll
  for (int dy = -4; dy <= 4; ++dy) {
    int yy = y + dy;
    if ((unsigned)yy < (unsigned)IH) {
      int o = yy * IW + x;
      sL  += rs[0 * total + o];
      sL2 += rs[1 * total + o];
      sR  += rs[2 * total + o];
      sR2 += rs[3 * total + o];
    }
  }
  st[0 * total + idx] = sL;
  st[1 * total + idx] = sqrtf(sL2 - sL * sL * (1.0f / NWIN) + 1e-5f);
  st[2 * total + idx] = sR;
  st[3 * total + idx] = sqrtf(sR2 - sR * sR * (1.0f / NWIN) + 1e-5f);
}

// ---------------------------------------------------------------------------
// Main cost-volume kernel. Grid: (W/16, ceil(H/56), 2 directions), 256 thr.
// 8 waves; wave w handles disparities d = w, w+8, ... independently (wave-
// private LDS scratch, no workgroup barriers inside the d-loop).
// 9-tap box filters run on the tensor pipe via banded-ones WMMA:
//   horizontal: D = q(16x32) x B, B[k][n]=1 iff n<=k<=n+8   (1 wmma / 16 rows)
//   vertical:   D = A x h(32x16), A[m][k]=1 iff m<=k<=m+8   (1 wmma / 16 rows)
// Output (249 MB, written once, never re-read) uses non-temporal stores so the
// streamed cost volume does not churn L2, keeping inputs + stat planes hot.
// ---------------------------------------------------------------------------
__global__ __launch_bounds__(256) void ncc_cost_kernel(const float* __restrict__ L,
                                                       const float* __restrict__ R,
                                                       const float* __restrict__ st,
                                                       float* __restrict__ out) {
  __shared__ __align__(16) float    fixT[ICH][RT][32];   // fixed image strip (cols x0-4..x0+19)
  __shared__ __align__(16) float    extT[ICH][RT][88];   // sliding image strip (83 cols used)
  __shared__ __align__(16) _Float16 qbuf[8][RT][32];     // per-wave pointwise products (cols 24..31 stay 0)
  __shared__ __align__(16) _Float16 hbuf[8][XT][80];     // per-wave row-boxed, transposed [x][y] (rows 64..79 stay 0)

  const int x0  = blockIdx.x * XT;
  const int y0  = blockIdx.y * YT;
  const int dir = blockIdx.z;            // 0 = costL, 1 = costR
  const int tid = threadIdx.x;
  const int HW  = IH * IW;

  const float* fimg = dir ? R : L;
  const float* mimg = dir ? L : R;
  const int fx0 = x0 - 4;
  const int ex0 = dir ? (x0 - 3) : (x0 - 64);
  const int ry0 = y0 - 4;

  // Bulk-zero the wave scratch once (b128 stores); guarantees the padded
  // regions read by WMMA fragments are 0.0 (never NaN garbage).
  {
    v4u z = {};
    v4u* q4 = (v4u*)&qbuf[0][0][0];
    for (int t = tid; t < (int)(sizeof(qbuf) / 16); t += 256) q4[t] = z;
    v4u* h4 = (v4u*)&hbuf[0][0][0];
    for (int t = tid; t < (int)(sizeof(hbuf) / 16); t += 256) h4[t] = z;
  }

  // Cooperative tile loads, zero padded (inputs live in L2 across the grid).
  for (int t = tid; t < ICH * RT * 24; t += 256) {
    int c = t / (RT * 24); int rm = t - c * (RT * 24);
    int h = rm / 24;       int j  = rm - h * 24;
    int y = ry0 + h, x = fx0 + j;
    float v = 0.f;
    if ((unsigned)y < (unsigned)IH && (unsigned)x < (unsigned)IW)
      v = fimg[(c * IH + y) * IW + x];
    fixT[c][h][j] = v;
  }
  for (int t = tid; t < ICH * RT * 83; t += 256) {
    int c = t / (RT * 83); int rm = t - c * (RT * 83);
    int h = rm / 83;       int j  = rm - h * 83;
    int y = ry0 + h, x = ex0 + j;
    float v = 0.f;
    if ((unsigned)y < (unsigned)IH && (unsigned)x < (unsigned)IW)
      v = mimg[(c * IH + y) * IW + x];
    extT[c][h][j] = v;
  }
  // Warm the fixed-column stat planes (global_prefetch_b8 path).
  if (tid < YT) {
    int y = y0 + tid;
    if (y < IH) {
      __builtin_prefetch(&st[(dir ? 2 : 0) * HW + y * IW + x0], 0, 0);
      __builtin_prefetch(&st[(dir ? 3 : 1) * HW + y * IW + x0], 0, 0);
    }
  }
  __syncthreads();

  const int wave = tid >> 5;
  const int lane = tid & 31;
  const int lm   = lane & 15;
  const int hi   = lane >> 4;

  // Banded-ones fragment (exact in f16). Under the symmetric 16-bit A/B
  // fragment layouts (lane<16: K in {0..7,16..23}; lane>=16: K in {8..15,24..31};
  // row/col index = lane%16) the same register pattern serves as the B operand
  // of the horizontal pass and the A operand of the vertical pass.
  v16h ones;
  #pragma unroll
  for (int e = 0; e < 16; ++e) {
    int k = (e < 8 ? e : e + 8) + hi * 8;
    ones[e] = (_Float16)((k >= lm && k <= lm + 8) ? 1.0f : 0.0f);
  }

  _Float16* qw = &qbuf[wave][0][0];
  _Float16* hw = &hbuf[wave][0][0];

  const int sf = dir ? 2 : 0;   // fixed-image stat plane base (Sum, then sqrt)
  const int sm = 2 - sf;        // moving-image stat plane base

  for (int d = wave; d < ND; d += 8) {
    const int i   = d + 1;
    const int off = dir ? d : (59 - d);   // ext column offset for this disparity

    // Pointwise cross product q[h][j] = sum_c fix[c][h][j] * ext[c][h][j+off]
    if (lane < 24) {
      const int j = lane;
      #pragma unroll 4
      for (int h = 0; h < RT; ++h) {
        float a = fixT[0][h][j] * extT[0][h][j + off]
                + fixT[1][h][j] * extT[1][h][j + off]
                + fixT[2][h][j] * extT[2][h][j + off];
        qw[h * 32 + j] = (_Float16)a;
      }
    }
    asm volatile("s_wait_dscnt 0" ::: "memory");

    // Horizontal 9-tap box sum: one v_wmma per 16-row group.
    #pragma unroll
    for (int r = 0; r < 4; ++r) {
      const _Float16* qrow = qw + (16 * r + lm) * 32;
      v8h a0 = *(const v8h*)(qrow + hi * 8);
      v8h a1 = *(const v8h*)(qrow + 16 + hi * 8);
      v16h A = __builtin_shufflevector(a0, a1, 0,1,2,3,4,5,6,7,8,9,10,11,12,13,14,15);
      v8f cz = {};
      v8f dte = __builtin_amdgcn_wmma_f32_16x16x32_f16(false, A, false, ones,
                                                       (short)0, cz, false, false);
      // D-fragment: lane holds col x=lm, rows 16r + 8*hi + v  -> contiguous
      // b128 store into the transposed hbuf[x][y].
      v8h p;
      #pragma unroll
      for (int v = 0; v < 8; ++v) p[v] = (_Float16)dte[v];
      *(v8h*)(hw + lm * 80 + 16 * r + hi * 8) = p;
    }
    asm volatile("s_wait_dscnt 0" ::: "memory");

    // Vertical 9-tap box sum + normalize + store: one v_wmma per 16-row group.
    #pragma unroll
    for (int g = 0; g < 4; ++g) {
      const _Float16* hcol = hw + lm * 80 + 16 * g;
      v8h b0 = *(const v8h*)(hcol + hi * 8);
      v8h b1 = *(const v8h*)(hcol + 16 + hi * 8);
      v16h B = __builtin_shufflevector(b0, b1, 0,1,2,3,4,5,6,7,8,9,10,11,12,13,14,15);
      v8f cz = {};
      v8f P = __builtin_amdgcn_wmma_f32_16x16x32_f16(false, ones, false, B,
                                                     (short)0, cz, false, false);

      const int x  = x0 + lm;
      const int xm = dir ? (x + i) : (x - i);
      const bool ok = dir ? (x < IW - i) : (x >= i);   // reference masking
      #pragma unroll
      for (int v = 0; v < 8; ++v) {
        int t = 16 * g + hi * 8 + v;
        int y = y0 + t;
        if (t < YT && y < IH) {
          float cost = 0.f;
          if (ok) {
            float Sfv = st[(sf    ) * HW + y * IW + x];
            float Qfv = st[(sf + 1) * HW + y * IW + x];
            float Smv = st[(sm    ) * HW + y * IW + xm];
            float Qmv = st[(sm + 1) * HW + y * IW + xm];
            float prod = P[v] - Sfv * Smv * (1.0f / NWIN);
            cost = (prod + CEPS) / (Qfv * Qmv + CEPS);
          }
          // Streamed once, never re-read: non-temporal store (TH=NT) keeps the
          // 249 MB cost volume from displacing inputs/stats in L2.
          __builtin_nontemporal_store(cost, &out[((dir * ND + d) * IH + y) * IW + x]);
        }
      }
    }
  }
}

// ---------------------------------------------------------------------------
extern "C" void kernel_launch(void* const* d_in, const int* in_sizes, int n_in,
                              void* d_out, int out_size, void* d_ws, size_t ws_size,
                              hipStream_t stream) {
  const float* L = (const float*)d_in[0];
  const float* R = (const float*)d_in[1];
  float* out = (float*)d_out;
  // Row-box temporaries live in d_out (overwritten later by the main kernel);
  // final stat planes {LSum, sqrtL, RSum, sqrtR} live in d_ws (8.3 MB).
  float* rs = out;
  float* st = (float*)d_ws;

  const int total = IH * IW;
  dim3 blk(256);
  dim3 g1((total + 255) / 256);
  stats_row_kernel<<<g1, blk, 0, stream>>>(L, R, rs);
  stats_col_kernel<<<g1, blk, 0, stream>>>(rs, st);

  dim3 g2(IW / XT, (IH + YT - 1) / YT, 2);
  ncc_cost_kernel<<<g2, blk, 0, stream>>>(L, R, st, out);
}
